// EncoderBlock_74354473828364
// MI455X (gfx1250) — matrix operand order repaired and checked by reading, not compiled, not run on hardware
//
#include <hip/hip_runtime.h>

// ---------------------------------------------------------------------------
// Transformer encoder block (pre-norm) for MI455X / gfx1250.
// GEMMs: v_wmma_f32_16x16x32_f16 + Tensor Data Mover (tensor_load_to_lds)
// double-buffered LDS staging. Attention: flash-style WMMA kernel.
// ---------------------------------------------------------------------------

typedef __attribute__((ext_vector_type(16))) _Float16 v16h;
typedef __attribute__((ext_vector_type(8)))  float    v8f;
typedef __attribute__((ext_vector_type(4)))  unsigned int u32x4;
typedef __attribute__((ext_vector_type(8)))  unsigned int u32x8;

union F16x16 {
    v16h v;
    uint4 q[2];
    _Float16 h[16];
};

#define D_MODEL 1024
#define SEQ     2048
#define NBATCH  2
#define NHEAD   16
#define DKH     64
#define DFF     4096
#define NTOK    (NBATCH * SEQ)   // 4096

// ---------------------------------------------------------------------------
// Tensor Data Mover: load a [rows x 32-half] tile (row stride = `stride`
// halves in global) into LDS at byte offset lds_off, padding 4 DWORDs after
// every 16 DWORDs so the LDS row stride is 40 halves (bank-conflict free).
// Issued by one wave; completion tracked with TENSORcnt.
// ---------------------------------------------------------------------------
__device__ __forceinline__ void tdm_load_tile128x32(const _Float16* gsrc,
                                                    unsigned lds_off,
                                                    unsigned rows,
                                                    unsigned stride_halves) {
    unsigned long long ga = (unsigned long long)(uintptr_t)gsrc;
    u32x4 g0;
    g0[0] = 1u;                                              // count=1 (valid D#)
    g0[1] = lds_off;                                         // lds_addr (bytes)
    g0[2] = (unsigned)(ga & 0xFFFFFFFFu);                    // global_addr[31:0]
    g0[3] = (unsigned)((ga >> 32) & 0x01FFFFFFu)             // global_addr[56:32]
          | 0x80000000u;                                     // type=2 ("image")
    u32x8 g1;
    g1[0] = (1u << 16)      // data_size = 2 bytes
          | (1u << 20)      // pad_enable
          | (3u << 22)      // pad_interval: 16 DWORDs
          | (3u << 25);     // pad_amount:   4 DWORDs  -> 40-half LDS row stride
    g1[1] = (stride_halves & 0xFFFFu) << 16;                 // tensor_dim0 lo16
    g1[2] = (stride_halves >> 16) | ((rows & 0xFFFFu) << 16);// dim0 hi / dim1 lo
    g1[3] = (rows >> 16) | (32u << 16);                      // dim1 hi / tile_dim0=32
    g1[4] = rows;                                            // tile_dim1 ; tile_dim2=0
    g1[5] = stride_halves;                                   // tensor_dim0_stride lo32
    g1[6] = 0u;                                              // stride hi / dim1_stride lo
    g1[7] = 0u;
    asm volatile("tensor_load_to_lds %0, %1" :: "s"(g0), "s"(g1) : "memory");
}

__device__ __forceinline__ unsigned lds_byte_off(const void* p) {
    return (unsigned)(uintptr_t)p;   // low 32 bits of LDS aperture addr = LDS offset
}

// ---------------------------------------------------------------------------
// fp32 -> f16 weight conversion (n multiple of 4)
// ---------------------------------------------------------------------------
__global__ __launch_bounds__(256) void cvt_f32_f16(const float* __restrict__ in,
                                                   _Float16* __restrict__ out, int n) {
    int i = (blockIdx.x * 256 + threadIdx.x) * 4;
    if (i < n) {
        float4 v = *reinterpret_cast<const float4*>(in + i);
        out[i + 0] = (_Float16)v.x;
        out[i + 1] = (_Float16)v.y;
        out[i + 2] = (_Float16)v.z;
        out[i + 3] = (_Float16)v.w;
    }
}

// ---------------------------------------------------------------------------
// LayerNorm (faithful: unbiased std, denom = sqrt(std + eps)), f16 output.
// ---------------------------------------------------------------------------
__global__ __launch_bounds__(256) void ln_kernel(const float* __restrict__ x,
                                                 const float* __restrict__ alpha,
                                                 const float* __restrict__ beta,
                                                 _Float16* __restrict__ out) {
    __shared__ float redS[8], redQ[8];
    const int row = blockIdx.x;
    const float* xr = x + (size_t)row * D_MODEL;
    float s = 0.f, ss = 0.f;
    for (int i = threadIdx.x; i < D_MODEL; i += 256) {
        float v = xr[i];
        s += v;
        ss += v * v;
    }
#pragma unroll
    for (int off = 16; off; off >>= 1) {
        s  += __shfl_xor(s, off);
        ss += __shfl_xor(ss, off);
    }
    const int wid = threadIdx.x >> 5, lane = threadIdx.x & 31;
    if (lane == 0) { redS[wid] = s; redQ[wid] = ss; }
    __syncthreads();
    float S = 0.f, Q = 0.f;
#pragma unroll
    for (int w = 0; w < 8; ++w) { S += redS[w]; Q += redQ[w]; }
    const float mean = S * (1.0f / D_MODEL);
    float var = (Q - (float)D_MODEL * mean * mean) * (1.0f / (D_MODEL - 1));
    var = var > 0.f ? var : 0.f;
    const float stdv = sqrtf(var);
    const float inv  = alpha[0] * rsqrtf(stdv + 1e-6f);
    const float b    = beta[0];
    for (int i = threadIdx.x; i < D_MODEL; i += 256)
        out[(size_t)row * D_MODEL + i] = (_Float16)((xr[i] - mean) * inv + b);
}

// ---------------------------------------------------------------------------
// WMMA GEMM: out[M,N] = act(A[M,K] * Bw[N,K]^T + bias[N]) (+ resid[M,N])
// Block tile 128x128, K-step 32, 8 waves; wave tile 32x64 (2x4 wmma tiles).
// Tiles staged by the Tensor Data Mover with double-buffered LDS:
// wave 0 issues next-tile TDM descriptors, then waits tensorcnt<=2 so the
// current tile (the two oldest in-order TDM ops) is resident while the next
// tile's DMA overlaps the WMMA work.
// ---------------------------------------------------------------------------
template <bool RELU, bool RESID, bool OUT16>
__global__ __launch_bounds__(256) void gemm_wmma(const _Float16* __restrict__ A,
                                                 const _Float16* __restrict__ Bw,
                                                 const float* __restrict__ bias,
                                                 const float* __restrict__ resid,
                                                 _Float16* __restrict__ out16,
                                                 float* __restrict__ out32,
                                                 int M, int N, int K) {
    __shared__ _Float16 As[2][128 * 40];   // 40-half rows (TDM pad inserts +8)
    __shared__ _Float16 Bs[2][128 * 40];
    const int tid  = threadIdx.x;
    const int lane = tid & 31;
    const int wid  = tid >> 5;
    const int l16  = lane & 15;
    const int lhi  = lane >> 4;
    const int wm   = (wid & 3) * 32;   // wave M offset inside block tile
    const int wn   = (wid >> 2) * 64;  // wave N offset inside block tile
    const int bm   = blockIdx.y * 128;
    const int bn   = blockIdx.x * 128;

    const _Float16* Abase = A  + (size_t)bm * K;
    const _Float16* Bbase = Bw + (size_t)bn * K;

    v8f acc[2][4] = {};

    const int nk = K >> 5;   // K / 32
    if (wid == 0) {          // prologue DMA: tile 0 -> buffer 0
        tdm_load_tile128x32(Abase, lds_byte_off(&As[0][0]), 128u, (unsigned)K);
        tdm_load_tile128x32(Bbase, lds_byte_off(&Bs[0][0]), 128u, (unsigned)K);
    }

    for (int i = 0; i < nk; ++i) {
        const int cur = i & 1;
        if (wid == 0) {
            if (i + 1 < nk) {   // prefetch next tile into the other buffer
                const int k1 = (i + 1) << 5;
                tdm_load_tile128x32(Abase + k1, lds_byte_off(&As[cur ^ 1][0]),
                                    128u, (unsigned)K);
                tdm_load_tile128x32(Bbase + k1, lds_byte_off(&Bs[cur ^ 1][0]),
                                    128u, (unsigned)K);
                __builtin_amdgcn_s_wait_tensorcnt(2);   // current tile resident
            } else {
                __builtin_amdgcn_s_wait_tensorcnt(0);
            }
        }
        __syncthreads();

        // ---- fragments (ISA VGPR layouts, wave32)
        F16x16 af[2], bf[4];
#pragma unroll
        for (int mt = 0; mt < 2; ++mt) {
            const _Float16* p = &As[cur][(wm + mt * 16 + l16) * 40 + lhi * 8];
            af[mt].q[0] = *reinterpret_cast<const uint4*>(p);        // K = 8*hi + 0..7
            af[mt].q[1] = *reinterpret_cast<const uint4*>(p + 16);   // K = 16+8*hi+0..7
        }
#pragma unroll
        for (int nt = 0; nt < 4; ++nt) {
            const _Float16* p = &Bs[cur][(wn + nt * 16 + l16) * 40 + lhi * 8];
            bf[nt].q[0] = *reinterpret_cast<const uint4*>(p);        // K = 8*hi + 0..7
            bf[nt].q[1] = *reinterpret_cast<const uint4*>(p + 16);   // K = 16+8*hi+0..7
        }
#pragma unroll
        for (int mt = 0; mt < 2; ++mt)
#pragma unroll
            for (int nt = 0; nt < 4; ++nt)
                acc[mt][nt] = __builtin_amdgcn_wmma_f32_16x16x32_f16(
                    false, af[mt].v, false, bf[nt].v, (short)0, acc[mt][nt], false, false);
        __syncthreads();   // all reads of buffer `cur` done before it is re-filled
    }

    // ---- epilogue: C layout is M = 8*hi + r, N = lane&15 per 16x16 tile
#pragma unroll
    for (int mt = 0; mt < 2; ++mt)
#pragma unroll
        for (int nt = 0; nt < 4; ++nt) {
            const int col = bn + wn + nt * 16 + l16;
            const float bv = bias[col];
#pragma unroll
            for (int r = 0; r < 8; ++r) {
                const int row = bm + wm + mt * 16 + lhi * 8 + r;
                float v = acc[mt][nt][r] + bv;
                if (RELU) v = v > 0.f ? v : 0.f;
                if (RESID) v += resid[(size_t)row * N + col];
                if (OUT16) out16[(size_t)row * N + col] = (_Float16)v;
                else       out32[(size_t)row * N + col] = v;
            }
        }
}

// ---------------------------------------------------------------------------
// Flash attention: grid = (SEQ/128, B*H), block = 256 (8 waves).
// Each wave owns 16 query rows; streams 64-key chunks through LDS.
// Q/K/V/ctx are f16 [B*S, D_MODEL] with head h at column h*64.
// ---------------------------------------------------------------------------
__global__ __launch_bounds__(256) void attn_kernel(const _Float16* __restrict__ Qm,
                                                   const _Float16* __restrict__ Km,
                                                   const _Float16* __restrict__ Vm,
                                                   const int* __restrict__ mask,
                                                   _Float16* __restrict__ Ctx) {
    __shared__ _Float16 Ks[64 * 72];          // [key][d]   (+8 pad)
    __shared__ _Float16 Vt[64 * 72];          // [d][key]   (+8 pad)
    __shared__ _Float16 Ps[8 * 16 * 72];      // per-wave P scratch [16][72]
    const int tid  = threadIdx.x;
    const int lane = tid & 31;
    const int wid  = tid >> 5;
    const int l16  = lane & 15;
    const int lhi  = lane >> 4;
    const int bh = blockIdx.y;
    const int b  = bh >> 4;           // NHEAD = 16
    const int h  = bh & 15;
    const int qrow0 = blockIdx.x * 128 + wid * 16;   // wave's query base (in S)

    const _Float16* qb = Qm + (size_t)b * SEQ * D_MODEL + (size_t)h * DKH;
    const _Float16* kb = Km + (size_t)b * SEQ * D_MODEL + (size_t)h * DKH;
    const _Float16* vb = Vm + (size_t)b * SEQ * D_MODEL + (size_t)h * DKH;

    // resident Q fragments (A layout): aq[kt] covers d = kt*32 .. kt*32+31
    F16x16 aq[2];
#pragma unroll
    for (int kt = 0; kt < 2; ++kt) {
        const _Float16* p = qb + (size_t)(qrow0 + l16) * D_MODEL + kt * 32 + lhi * 8;
        aq[kt].q[0] = *reinterpret_cast<const uint4*>(p);
        aq[kt].q[1] = *reinterpret_cast<const uint4*>(p + 16);
    }

    v8f ctx[4] = {};
    float rmax[8], rsum[8];
#pragma unroll
    for (int r = 0; r < 8; ++r) { rmax[r] = -1e30f; rsum[r] = 0.f; }
    const float scale = 0.125f;   // 1/sqrt(64)

    for (int kc = 0; kc < SEQ; kc += 64) {
        __syncthreads();   // previous chunk fully consumed
        // ---- stage K chunk [64 keys][64 d]
#pragma unroll
        for (int i = 0; i < 2; ++i) {
            const int idx = tid + i * 256;        // 0..511 uint4
            const int key = idx >> 3;             // 8 uint4 per 64-half row
            const int c4  = idx & 7;
            uint4 d = *reinterpret_cast<const uint4*>(kb + (size_t)(kc + key) * D_MODEL + c4 * 8);
            *reinterpret_cast<uint4*>(&Ks[key * 72 + c4 * 8]) = d;
        }
        // ---- stage V chunk transposed: Vt[d][key]
        {
            const int key = tid & 63;
            const int d0  = (tid >> 6) * 16;
            const _Float16* vp = vb + (size_t)(kc + key) * D_MODEL + d0;
            _Float16 tmp[16];
            *reinterpret_cast<uint4*>(&tmp[0]) = *reinterpret_cast<const uint4*>(vp);
            *reinterpret_cast<uint4*>(&tmp[8]) = *reinterpret_cast<const uint4*>(vp + 8);
#pragma unroll
            for (int j = 0; j < 16; ++j) Vt[(d0 + j) * 72 + key] = tmp[j];
        }
        __syncthreads();

        // ---- scores S = Q K^T * scale (+mask), 4 tiles of 16 keys
        v8f s[4];
#pragma unroll
        for (int nt = 0; nt < 4; ++nt) {
            v8f a = {};
#pragma unroll
            for (int kt = 0; kt < 2; ++kt) {
                F16x16 bf;
                const _Float16* p = &Ks[(nt * 16 + l16) * 72 + kt * 32 + lhi * 8];
                bf.q[0] = *reinterpret_cast<const uint4*>(p);
                bf.q[1] = *reinterpret_cast<const uint4*>(p + 16);
                a = __builtin_amdgcn_wmma_f32_16x16x32_f16(false, aq[kt].v, false, bf.v,
                                                           (short)0, a, false, false);
            }
            const int key = kc + nt * 16 + l16;
            const float madd = (mask[(size_t)b * SEQ + key] == 0) ? -1e9f : 0.f;
#pragma unroll
            for (int r = 0; r < 8; ++r) s[nt][r] = a[r] * scale + madd;
        }

        // ---- online softmax (rows striped: row = 8*hi + r, 16 lanes per row)
        float tmax[8];
#pragma unroll
        for (int r = 0; r < 8; ++r) {
            float m = s[0][r];
#pragma unroll
            for (int nt = 1; nt < 4; ++nt) m = fmaxf(m, s[nt][r]);
            tmax[r] = m;
        }
#pragma unroll
        for (int off = 8; off; off >>= 1)
#pragma unroll
            for (int r = 0; r < 8; ++r) tmax[r] = fmaxf(tmax[r], __shfl_xor(tmax[r], off));
        float nscale[8];
#pragma unroll
        for (int r = 0; r < 8; ++r) {
            const float nm = fmaxf(rmax[r], tmax[r]);
            nscale[r] = __expf(rmax[r] - nm);
            rmax[r] = nm;
        }
        float tsum[8] = {};
#pragma unroll
        for (int nt = 0; nt < 4; ++nt)
#pragma unroll
            for (int r = 0; r < 8; ++r) {
                const float p = __expf(s[nt][r] - rmax[r]);
                s[nt][r] = p;
                tsum[r] += p;
            }
#pragma unroll
        for (int off = 8; off; off >>= 1)
#pragma unroll
            for (int r = 0; r < 8; ++r) tsum[r] += __shfl_xor(tsum[r], off);
#pragma unroll
        for (int r = 0; r < 8; ++r) rsum[r] = rsum[r] * nscale[r] + tsum[r];
#pragma unroll
        for (int dt = 0; dt < 4; ++dt)
#pragma unroll
            for (int r = 0; r < 8; ++r) ctx[dt][r] *= nscale[r];

        // ---- restripe P (C layout -> A layout) through per-wave LDS
        _Float16* ps = &Ps[wid * 16 * 72];
#pragma unroll
        for (int nt = 0; nt < 4; ++nt)
#pragma unroll
            for (int r = 0; r < 8; ++r)
                ps[(lhi * 8 + r) * 72 + nt * 16 + l16] = (_Float16)s[nt][r];
        asm volatile("" ::: "memory");   // same-wave LDS is in-order; stop reordering
        F16x16 pa[2];
#pragma unroll
        for (int kt = 0; kt < 2; ++kt) {
            const _Float16* p = &ps[l16 * 72 + kt * 32 + lhi * 8];
            pa[kt].q[0] = *reinterpret_cast<const uint4*>(p);
            pa[kt].q[1] = *reinterpret_cast<const uint4*>(p + 16);
        }

        // ---- ctx += P V  (4 dk tiles x 2 key-halves)
#pragma unroll
        for (int dt = 0; dt < 4; ++dt)
#pragma unroll
            for (int kt = 0; kt < 2; ++kt) {
                F16x16 bf;
                const _Float16* p = &Vt[(dt * 16 + l16) * 72 + kt * 32 + lhi * 8];
                bf.q[0] = *reinterpret_cast<const uint4*>(p);
                bf.q[1] = *reinterpret_cast<const uint4*>(p + 16);
                ctx[dt] = __builtin_amdgcn_wmma_f32_16x16x32_f16(false, pa[kt].v, false, bf.v,
                                                                 (short)0, ctx[dt], false, false);
            }
    }

    // ---- normalize + store ctx (f16, [token, h*64+d])
#pragma unroll
    for (int dt = 0; dt < 4; ++dt)
#pragma unroll
        for (int r = 0; r < 8; ++r) {
            const int qrow = qrow0 + lhi * 8 + r;
            const int d    = dt * 16 + l16;
            const float v  = ctx[dt][r] * (1.0f / rsum[r]);
            Ctx[((size_t)b * SEQ + qrow) * D_MODEL + h * DKH + d] = (_Float16)v;
        }
}

// ---------------------------------------------------------------------------
// Host launcher
// ---------------------------------------------------------------------------
extern "C" void kernel_launch(void* const* d_in, const int* in_sizes, int n_in,
                              void* d_out, int out_size, void* d_ws, size_t ws_size,
                              hipStream_t stream) {
    const float* x     = (const float*)d_in[0];
    const int*   mask  = (const int*)  d_in[1];
    const float* wq    = (const float*)d_in[2];
    const float* bq    = (const float*)d_in[3];
    const float* wk    = (const float*)d_in[4];
    const float* bk    = (const float*)d_in[5];
    const float* wv    = (const float*)d_in[6];
    const float* bv    = (const float*)d_in[7];
    const float* wo    = (const float*)d_in[8];
    const float* bo    = (const float*)d_in[9];
    const float* w1    = (const float*)d_in[10];
    const float* b1    = (const float*)d_in[11];
    const float* w2    = (const float*)d_in[12];
    const float* b2    = (const float*)d_in[13];
    const float* ln1a  = (const float*)d_in[14];
    const float* ln1b  = (const float*)d_in[15];
    const float* ln2a  = (const float*)d_in[16];
    const float* ln2b  = (const float*)d_in[17];
    float* out = (float*)d_out;

    const size_t MB = 1024ull * 1024ull;
    char* ws = (char*)d_ws;
    _Float16* wq16 = (_Float16*)(ws + 0 * MB);    // 2 MB
    _Float16* wk16 = (_Float16*)(ws + 2 * MB);    // 2 MB
    _Float16* wv16 = (_Float16*)(ws + 4 * MB);    // 2 MB
    _Float16* wo16 = (_Float16*)(ws + 6 * MB);    // 2 MB
    _Float16* w116 = (_Float16*)(ws + 8 * MB);    // 8 MB
    _Float16* w216 = (_Float16*)(ws + 16 * MB);   // 8 MB
    _Float16* xn   = (_Float16*)(ws + 24 * MB);   // 8 MB (reused for ln2 output)
    _Float16* qbuf = (_Float16*)(ws + 32 * MB);   // 8 MB
    _Float16* kbuf = (_Float16*)(ws + 40 * MB);   // 8 MB
    _Float16* vbuf = (_Float16*)(ws + 48 * MB);   // 8 MB
    _Float16* ctxb = (_Float16*)(ws + 56 * MB);   // 8 MB
    float*    x2   = (float*)   (ws + 64 * MB);   // 16 MB
    _Float16* h1   = (_Float16*)(ws + 80 * MB);   // 32 MB   (total 112 MB)

    const int nW  = D_MODEL * D_MODEL;   // 1M
    const int nW1 = DFF * D_MODEL;       // 4M

    // 1) weight conversion fp32 -> f16
    cvt_f32_f16<<<nW  / 1024, 256, 0, stream>>>(wq, wq16, nW);
    cvt_f32_f16<<<nW  / 1024, 256, 0, stream>>>(wk, wk16, nW);
    cvt_f32_f16<<<nW  / 1024, 256, 0, stream>>>(wv, wv16, nW);
    cvt_f32_f16<<<nW  / 1024, 256, 0, stream>>>(wo, wo16, nW);
    cvt_f32_f16<<<nW1 / 1024, 256, 0, stream>>>(w1, w116, nW1);
    cvt_f32_f16<<<nW1 / 1024, 256, 0, stream>>>(w2, w216, nW1);

    // 2) LN1
    ln_kernel<<<NTOK, 256, 0, stream>>>(x, ln1a, ln1b, xn);

    // 3) Q/K/V projections (WMMA + TDM)
    dim3 gP(D_MODEL / 128, NTOK / 128);     // (8, 32)
    gemm_wmma<false, false, true><<<gP, 256, 0, stream>>>(xn, wq16, bq, nullptr, qbuf, nullptr,
                                                          NTOK, D_MODEL, D_MODEL);
    gemm_wmma<false, false, true><<<gP, 256, 0, stream>>>(xn, wk16, bk, nullptr, kbuf, nullptr,
                                                          NTOK, D_MODEL, D_MODEL);
    gemm_wmma<false, false, true><<<gP, 256, 0, stream>>>(xn, wv16, bv, nullptr, vbuf, nullptr,
                                                          NTOK, D_MODEL, D_MODEL);

    // 4) attention
    dim3 gA(SEQ / 128, NBATCH * NHEAD);     // (16, 32)
    attn_kernel<<<gA, 256, 0, stream>>>(qbuf, kbuf, vbuf, mask, ctxb);

    // 5) output projection + residual -> x2 (fp32)
    gemm_wmma<false, true, false><<<gP, 256, 0, stream>>>(ctxb, wo16, bo, x, nullptr, x2,
                                                          NTOK, D_MODEL, D_MODEL);

    // 6) LN2 -> xn (reuse)
    ln_kernel<<<NTOK, 256, 0, stream>>>(x2, ln2a, ln2b, xn);

    // 7) FFN1 with ReLU -> h1 (f16)
    dim3 gF1(DFF / 128, NTOK / 128);        // (32, 32)
    gemm_wmma<true, false, true><<<gF1, 256, 0, stream>>>(xn, w116, b1, nullptr, h1, nullptr,
                                                          NTOK, DFF, D_MODEL);

    // 8) FFN2 + residual -> d_out (fp32)
    gemm_wmma<false, true, false><<<gP, 256, 0, stream>>>(h1, w216, b2, x2, nullptr, out,
                                                          NTOK, D_MODEL, DFF);
}
